// kNN_48395691491888
// MI455X (gfx1250) — compile-verified, matrix operand order checked by reading, not run on hardware
//
#include <hip/hip_runtime.h>

// ---------------- problem constants ----------------
#define NQ      2048
#define NT      100000
#define DIMS    128
#define TOPK    16
#define SPLITS  16
#define NQTILES (NQ / 16)                        // 128
#define NTILES  (NT / 16)                        // 6250 (exact)
#define TPS     ((NTILES + SPLITS - 1) / SPLITS) // 391 tiles per split
#define CAND_PER_Q (SPLITS * 2 * TOPK)           // 512 candidates per query
#define QT_PER_BLOCK 8                           // query tiles (waves) per block
#define TILE_BYTES 8192                          // one swizzled 16x128 bf16 hi+lo tile
#define NEG_BIG (-3.0e38f)

// ws layout (byte offsets):
//   [0 .. (NT+NQ)*4)        row squared norms (train rows then test rows), f32
//   [OFF_TRAIN_B ..)        swizzled train WMMA-A operands: NTILES * 8192 B
//   [OFF_QB_B   ..)         swizzled query WMMA-B operands: NQTILES * 8192 B
//   [OFF_CV_B   ..)         candidate keys   (float) NQ*512   (key = tn - 2*dot)
//   [OFF_CI_B   ..)         candidate indices (int32) NQ*512
#define OFF_TRAIN_B 409600ull
#define OFF_QB_B    (OFF_TRAIN_B + (unsigned long long)NTILES * TILE_BYTES)   // 51,609,600
#define OFF_CV_B    (OFF_QB_B + (unsigned long long)NQTILES * TILE_BYTES)     // 52,658,176
#define OFF_CI_B    (OFF_CV_B + (unsigned long long)NQ * CAND_PER_Q * 4)      // 56,852,480
// total ws needed: OFF_CI_B + NQ*512*4  ~= 61 MB

typedef __attribute__((ext_vector_type(16))) __bf16 v16bf;
typedef __attribute__((ext_vector_type(8)))  float  v8f;
typedef __attribute__((ext_vector_type(4)))  float  v4f;

// ---------------------------------------------------------------------------
// Kernel 1: squared row norms for x_train (rows 0..NT-1) and x_test (NT..)
// ---------------------------------------------------------------------------
__global__ void knn_norms_kernel(const float* __restrict__ xtest,
                                 const float* __restrict__ xtrain,
                                 float* __restrict__ norms) {
    int r = blockIdx.x * blockDim.x + threadIdx.x;
    if (r >= NT + NQ) return;
    const float* row = (r < NT) ? (xtrain + (size_t)r * DIMS)
                                : (xtest  + (size_t)(r - NT) * DIMS);
    const v4f* p = (const v4f*)row;
    float s = 0.0f;
#pragma unroll
    for (int i = 0; i < DIMS / 4; ++i) {
        v4f v = p[i];
        s += v[0]*v[0] + v[1]*v[1] + v[2]*v[2] + v[3]*v[3];
    }
    norms[r] = s;
}

// ---------------------------------------------------------------------------
// Kernel 2: pre-swizzle f32 rows into WMMA bf16 hi/lo operand layout.
// 16-bit A (16x32) per WMMA chunk w: lanes 0-15 hold row M=lane, elements 0..7
// are K=32w+0..7, elements 8..15 are K=32w+16..23; lanes 16-31 get the
// +8-shifted K ranges. Per (tile,w,lane) we emit two 32-byte v16bf chunks
// (hi then lo split: x = hi + lo, lo = bf16(x - float(hi))).
// Tile block layout (8192 B): offset = ((w*2 + h)*32 + lane) * 32.
// ---------------------------------------------------------------------------
__global__ void knn_swizzle_kernel(const float* __restrict__ xtest,
                                   const float* __restrict__ xtrain,
                                   unsigned char* __restrict__ wsb) {
    const int totalTrain = NTILES * 4 * 32;
    const int totalQ     = NQTILES * 4 * 32;
    int c = blockIdx.x * blockDim.x + threadIdx.x;
    if (c >= totalTrain + totalQ) return;

    const float* mat;
    unsigned char* dst;
    int tile, rem;
    if (c < totalTrain) {
        tile = c / 128; rem = c % 128;
        mat  = xtrain;
        dst  = wsb + OFF_TRAIN_B + (size_t)tile * TILE_BYTES;
    } else {
        int cq = c - totalTrain;
        tile = cq / 128; rem = cq % 128;
        mat  = xtest;
        dst  = wsb + OFF_QB_B + (size_t)tile * TILE_BYTES;
    }
    const int w    = rem / 32;
    const int lane = rem % 32;
    const int m    = lane & 15;
    const int hih  = lane >> 4;

    const float* row = mat + (size_t)(tile * 16 + m) * DIMS;
    const float* p1  = row + 32 * w + 8 * hih;
    const float* p2  = p1 + 16;

    v16bf vh, vl;
#pragma unroll
    for (int i = 0; i < 2; ++i) {
        v4f u1 = *(const v4f*)(p1 + 4 * i);
        v4f u2 = *(const v4f*)(p2 + 4 * i);
#pragma unroll
        for (int j = 0; j < 4; ++j) {
            float x1 = u1[j];
            __bf16 h1 = (__bf16)x1;
            vh[4 * i + j] = h1;
            vl[4 * i + j] = (__bf16)(x1 - (float)h1);
            float x2 = u2[j];
            __bf16 h2 = (__bf16)x2;
            vh[8 + 4 * i + j] = h2;
            vl[8 + 4 * i + j] = (__bf16)(x2 - (float)h2);
        }
    }
    *(v16bf*)(dst + ((size_t)((w * 2 + 0) * 32 + lane)) * 32) = vh;
    *(v16bf*)(dst + ((size_t)((w * 2 + 1) * 32 + lane)) * 32) = vl;
}

// sorted-descending insert of (d, idx) into a 16-entry register list
__device__ __forceinline__ void topk_insert(float d, int idx,
                                            float (&vals)[TOPK], int (&idxs)[TOPK]) {
    if (d > vals[TOPK - 1]) {
        float cv = d; int ci = idx;
#pragma unroll
        for (int j = 0; j < TOPK; ++j) {
            if (cv > vals[j]) {
                float tv = vals[j]; int ti = idxs[j];
                vals[j] = cv; idxs[j] = ci;
                cv = tv; ci = ti;
            }
        }
    }
}

// CDNA5 async copy: global -> LDS, 16 bytes, tracked by ASYNCcnt.
__device__ __forceinline__ void async_g2lds_b128(unsigned lds_addr,
                                                 unsigned long long sbase,
                                                 unsigned voff) {
    asm volatile("global_load_async_to_lds_b128 %0, %1, %2"
                 : : "v"(lds_addr), "v"(voff), "s"(sbase) : "memory");
}
__device__ __forceinline__ void wait_asynccnt0() {
    asm volatile("s_wait_asynccnt 0x0" : : : "memory");
}

// ---------------------------------------------------------------------------
// Kernel 3: fused distances + per-lane top-16 on the monotone key
// key = tn - 2*dot  (qn and sqrt are constant/monotone per lane -> applied in
// the merge kernel; keeps the hot loop at 1 FMA + 1 compare per candidate).
// grid = (SPLITS, NQTILES/8), block = 256 (8 waves; wave wv owns query tile
// blockIdx.y*8+wv). Train tile operand block (8 KB) staged into LDS once per
// block with global_load_async_to_lds_b128, double buffered. Each lane holds
// query n = lane&15 and 8 train rows m = v + 8*(lane>>4) of the accumulator.
// ---------------------------------------------------------------------------
__global__ void __launch_bounds__(256)
knn_tile_kernel(const float* __restrict__ norms,
                const unsigned char* __restrict__ trainA,
                const unsigned char* __restrict__ queryB,
                float* __restrict__ cv, int* __restrict__ ci) {
    __shared__ __align__(1024) unsigned char smem[2 * TILE_BYTES];

    const int tid   = threadIdx.x;
    const int lane  = tid & 31;
    const int wv    = tid >> 5;
    const int m     = lane & 15;
    const int hih   = lane >> 4;
    const int split = blockIdx.x;
    const int qt    = blockIdx.y * QT_PER_BLOCK + wv;
    const int qbase = qt * 16;

    // per-wave query operand (B): hi/lo split, 4 K-chunks (64 VGPRs)
    v16bf bqh[4], bql[4];
    {
        const unsigned char* qblk = queryB + (size_t)qt * TILE_BYTES;
#pragma unroll
        for (int w = 0; w < 4; ++w) {
            bqh[w] = *(const v16bf*)(qblk + ((size_t)((w * 2 + 0) * 32 + lane)) * 32);
            bql[w] = *(const v16bf*)(qblk + ((size_t)((w * 2 + 1) * 32 + lane)) * 32);
        }
    }

    float vals[TOPK]; int idxs[TOPK];
#pragma unroll
    for (int j = 0; j < TOPK; ++j) { vals[j] = NEG_BIG; idxs[j] = 0; }

    const int t0 = split * TPS;
    const int t1 = (t0 + TPS < NTILES) ? (t0 + TPS) : NTILES;

    const unsigned lds0 = (unsigned)(size_t)(&smem[0]);
    const unsigned long long tbase = (unsigned long long)(size_t)trainA;

    // prologue: stage first tile into buffer 0 (each thread copies 32 B)
    {
        unsigned voff = (unsigned)((size_t)t0 * TILE_BYTES + tid * 32);
        unsigned dsta = lds0 + tid * 32;
        async_g2lds_b128(dsta, tbase, voff);
        async_g2lds_b128(dsta + 16, tbase, voff + 16);
    }

    for (int t = t0; t < t1; ++t) {
        const int p = (t - t0) & 1;
        wait_asynccnt0();
        __syncthreads();

        // overlap: kick off copy of next tile into the other buffer
        if (t + 1 < t1) {
            unsigned voff = (unsigned)((size_t)(t + 1) * TILE_BYTES + tid * 32);
            unsigned dsta = lds0 + (1 - p) * TILE_BYTES + tid * 32;
            async_g2lds_b128(dsta, tbase, voff);
            async_g2lds_b128(dsta + 16, tbase, voff + 16);
        }

        const unsigned char* buf = smem + p * TILE_BYTES;
        v8f acc = {};
#pragma unroll
        for (int w = 0; w < 4; ++w) {
            v16bf ah = *(const v16bf*)(buf + ((size_t)((w * 2 + 0) * 32 + lane)) * 32);
            v16bf al = *(const v16bf*)(buf + ((size_t)((w * 2 + 1) * 32 + lane)) * 32);
            // dot ~= ah*bh + ah*bl + al*bh (fp32 accumulate)
            acc = __builtin_amdgcn_wmma_f32_16x16x32_bf16(
                      false, ah, false, bqh[w], (short)0, acc, false, false);
            acc = __builtin_amdgcn_wmma_f32_16x16x32_bf16(
                      false, ah, false, bql[w], (short)0, acc, false, false);
            acc = __builtin_amdgcn_wmma_f32_16x16x32_bf16(
                      false, al, false, bqh[w], (short)0, acc, false, false);
        }

        const int nb = t * 16;
        const float* tp = norms + nb + 8 * hih;   // this lane's 8 train norms
        v4f tn0 = *(const v4f*)(tp);
        v4f tn1 = *(const v4f*)(tp + 4);
#pragma unroll
        for (int v = 0; v < 8; ++v) {
            float tn  = (v < 4) ? tn0[v & 3] : tn1[v & 3];
            float key = fmaf(-2.0f, acc[v], tn);   // tn - 2*dot (monotone in dist)
            topk_insert(key, nb + 8 * hih + v, vals, idxs);
        }
    }

    // emit this lane's 16 candidates: cand[q][split][hih][j]
    const int q = qbase + m;
    size_t base = (((size_t)q * SPLITS + split) * 2 + hih) * TOPK;
#pragma unroll
    for (int j = 0; j < TOPK; ++j) {
        cv[base + j] = vals[j];
        ci[base + j] = idxs[j];
    }
}

// ---------------------------------------------------------------------------
// Kernel 4: merge 512 candidates per query -> final top-16 (desc, stable
// ties); apply + qn and sqrt only to the 16 winners.
// ---------------------------------------------------------------------------
__global__ void knn_merge_kernel(const float* __restrict__ norms,
                                 const float* __restrict__ cv,
                                 const int* __restrict__ ci,
                                 float* __restrict__ out_vals,
                                 int* __restrict__ out_idx) {
    int q = blockIdx.x * blockDim.x + threadIdx.x;
    if (q >= NQ) return;
    const float* qv = cv + (size_t)q * CAND_PER_Q;
    const int*   qi = ci + (size_t)q * CAND_PER_Q;

    float vals[TOPK]; int idxs[TOPK];
#pragma unroll
    for (int j = 0; j < TOPK; ++j) { vals[j] = NEG_BIG; idxs[j] = 0; }

    for (int c = 0; c < CAND_PER_Q; ++c)
        topk_insert(qv[c], qi[c], vals, idxs);

    const float qn = norms[NT + q];
#pragma unroll
    for (int j = 0; j < TOPK; ++j) {
        out_vals[q * TOPK + j] = sqrtf(fmaxf(vals[j] + qn, 0.0f));
        out_idx [q * TOPK + j] = idxs[j];
    }
}

// ---------------------------------------------------------------------------
extern "C" void kernel_launch(void* const* d_in, const int* in_sizes, int n_in,
                              void* d_out, int out_size, void* d_ws, size_t ws_size,
                              hipStream_t stream) {
    const float* xtest  = (const float*)d_in[0];  // [2048, 128] f32
    const float* xtrain = (const float*)d_in[1];  // [100000, 128] f32

    unsigned char* wsb = (unsigned char*)d_ws;
    float* norms  = (float*)d_ws;
    const unsigned char* trainA = wsb + OFF_TRAIN_B;
    const unsigned char* queryB = wsb + OFF_QB_B;
    float* cv = (float*)(wsb + OFF_CV_B);
    int*   ci = (int*)(wsb + OFF_CI_B);

    float* out_vals = (float*)d_out;              // [2048, 16] f32
    int*   out_idx  = (int*)d_out + NQ * TOPK;    // [2048, 16] i32

    // 1) row squared norms
    knn_norms_kernel<<<(NT + NQ + 255) / 256, 256, 0, stream>>>(xtest, xtrain, norms);

    // 2) one-time bf16 hi/lo operand swizzle (no conversion VALU in hot loop)
    {
        int chunks = (NTILES + NQTILES) * 4 * 32;
        knn_swizzle_kernel<<<(chunks + 255) / 256, 256, 0, stream>>>(xtest, xtrain, wsb);
    }

    // 3) fused WMMA distances + per-lane top-16 candidates (on monotone key)
    dim3 grid(SPLITS, NQTILES / QT_PER_BLOCK);
    knn_tile_kernel<<<grid, 256, 0, stream>>>(norms, trainA, queryB, cv, ci);

    // 4) merge candidates -> final top-16 per query (adds qn, takes sqrt)
    knn_merge_kernel<<<(NQ + 255) / 256, 256, 0, stream>>>(norms, cv, ci, out_vals, out_idx);
}
